// SelfAttention_35691178230212
// MI455X (gfx1250) — compile-verified, matrix operand order checked by reading, not run on hardware
//
#include <hip/hip_runtime.h>
#include <hip/hip_bf16.h>
#include <math.h>

// ---------------------------------------------------------------------------
// CDNA5 (gfx1250) wave32 WMMA types + helpers
// ---------------------------------------------------------------------------
typedef __attribute__((ext_vector_type(16))) __bf16 v16bf;
typedef __attribute__((ext_vector_type(8)))  float  v8f;
typedef unsigned int u32x4 __attribute__((ext_vector_type(4)));
typedef int          i32x4 __attribute__((ext_vector_type(4)));
typedef int          i32x8 __attribute__((ext_vector_type(8)));

#define NPTS   4096
#define BATCH  8
#define CDIM   64
#define KNN    20
#define NEG_INF (-3.4e38f)

#if defined(__has_builtin)
#if __has_builtin(__builtin_amdgcn_tensor_load_to_lds)
#define HAVE_TDM 1
#else
#define HAVE_TDM 0
#endif
#else
#define HAVE_TDM 0
#endif

__device__ __forceinline__ v8f wmma_bf16(v16bf a, v16bf b, v8f c) {
  // D = A(16x32 bf16) x B(32x16 bf16) + C(16x16 f32)
  return __builtin_amdgcn_wmma_f32_16x16x32_bf16(
      /*neg_a=*/false, a, /*neg_b=*/false, b,
      /*c_mod=*/(short)0, c, /*reuse_a=*/false, /*reuse_b=*/false);
}

// Pack one lane's A (or B) fragment for a 16x32 bf16 tile from a row-major
// 'row' of bf16 (row = the lane's M-row for A, or N-column source for B).
// ISA 16-bit A 16x32 layout: lanes 0-15 hold K 0..7 & 16..23, lanes 16-31
// hold K 8..15 & 24..31. Chunks are contiguous 16B pairs -> b128 loads.
__device__ __forceinline__ v16bf load_tile(const __bf16* row, int lane, int kc) {
  int kb = kc * 32 + ((lane & 16) ? 8 : 0);
  v16bf r;
#pragma unroll
  for (int e = 0; e < 16; ++e) {
    int k = kb + (e & 7) + ((e & 8) ? 16 : 0);
    r[e] = row[k];
  }
  return r;
}

#if HAVE_TDM
// 1-D contiguous TDM copy global -> LDS, 'bytes' must be a multiple of 8.
// D# group0: count=1, lds byte addr, 57-bit global addr, type=2 ("image").
// D# group1: data_size=8B, tensor_dim0 = tile_dim0 = bytes/8 units,
//            tensor_dim0_stride = units, tensor_dim1 = 1, everything else 0.
// Executed by ONE wave; waits TENSORcnt==0 before returning.
__device__ __forceinline__ void tdm_load_1d(void* lds_dst, const void* gsrc,
                                            unsigned bytes) {
  unsigned units = bytes >> 3;  // 8-byte elements (tile_dim0 fits in 16 bits)
  unsigned lds_addr = (unsigned)(size_t)lds_dst;      // addr[31:0] = LDS offset
  unsigned long long ga = (unsigned long long)(size_t)gsrc;
  u32x4 g0;
  g0[0] = 1u;                                          // count=1, user mode
  g0[1] = lds_addr;                                    // lds_addr
  g0[2] = (unsigned)ga;                                // global_addr[31:0]
  g0[3] = (unsigned)((ga >> 32) & 0x01FFFFFFu) | 0x80000000u; // [56:32]|type=2
  i32x8 g1;
  g1[0] = (int)(3u << 16);                             // data_size=3 (8B)
  g1[1] = (int)((units & 0xFFFFu) << 16);              // tensor_dim0[15:0]
  g1[2] = (int)(((units >> 16) & 0xFFFFu) | (1u << 16)); // dim0 hi | dim1=1
  g1[3] = (int)((units & 0xFFFFu) << 16);              // tile_dim0
  g1[4] = 0;                                           // tile_dim1/2 unused
  g1[5] = (int)units;                                  // tensor_dim0_stride lo
  g1[6] = 0;
  g1[7] = 0;
  i32x4 z4 = {};
#if __clang_major__ >= 23
  i32x8 z8 = {};
  __builtin_amdgcn_tensor_load_to_lds(g0, g1, z4, z4, z8, 0);
#else
  __builtin_amdgcn_tensor_load_to_lds(g0, g1, z4, z4, 0);
#endif
  __builtin_amdgcn_s_wait_tensorcnt((short)0);
}
#endif

// ---------------------------------------------------------------------------
// K0: transpose x (B,C,N) -> pts (B,N,C) in f32 + bf16, and squared norms
// ---------------------------------------------------------------------------
__global__ __launch_bounds__(256) void prep_points(
    const float* __restrict__ x, float* __restrict__ ptsF,
    __bf16* __restrict__ ptsH, float* __restrict__ sq) {
  int p = blockIdx.x * 256 + threadIdx.x;           // (b,n) flat, B*N
  if (p >= BATCH * NPTS) return;
  int b = p >> 12, n = p & (NPTS - 1);
  const float* xb = x + (size_t)b * CDIM * NPTS;
  float s = 0.f;
#pragma unroll
  for (int c = 0; c < CDIM; ++c) {
    float v = xb[(size_t)c * NPTS + n];
    ptsF[(size_t)p * CDIM + c] = v;
    ptsH[(size_t)p * CDIM + c] = (__bf16)v;
    s += v * v;
  }
  sq[p] = s;
}

// ---------------------------------------------------------------------------
// K0b: convert Wk, Wv, Wout to bf16
// ---------------------------------------------------------------------------
__global__ __launch_bounds__(256) void conv_weights(
    const float* __restrict__ Wk, const float* __restrict__ Wv,
    const float* __restrict__ Wout, __bf16* __restrict__ WkH,
    __bf16* __restrict__ WvH, __bf16* __restrict__ WoutH) {
  int i = blockIdx.x * 256 + threadIdx.x;
  if (i < 4 * 64 * 64) { WkH[i] = (__bf16)Wk[i]; WvH[i] = (__bf16)Wv[i]; }
  if (i < 128 * 256)   { WoutH[i] = (__bf16)Wout[i]; }
}

// ---------------------------------------------------------------------------
// K1: KNN. One workgroup = 16 rows of one batch. 8 waves each compute one
// 16x16 inner-product tile per iteration via bf16 WMMA; 256 threads keep
// register top-20 lists (16 slots per row); merged per row at the end.
// pdist(m,j) = 2<p_m,p_j> - |p_m|^2 - |p_j|^2  (larger = nearer)
// ---------------------------------------------------------------------------
__global__ __launch_bounds__(256) void knn_kernel(
    const __bf16* __restrict__ ptsH, const float* __restrict__ sq,
    int* __restrict__ idxOut) {
  __shared__ __bf16 aT[16 * CDIM];
  __shared__ float  sqRow[16];
  __shared__ float  tileBuf[8][16][16];
  __shared__ float  candV[16][16 * KNN];
  __shared__ int    candI[16][16 * KNN];

  int b  = blockIdx.x >> 8;              // 256 row-tiles per batch
  int r0 = (blockIdx.x & 255) * 16;
  int t = threadIdx.x, lane = t & 31, wave = t >> 5;

#if HAVE_TDM
  if (t < 32)
    tdm_load_1d(aT, ptsH + ((size_t)(b * NPTS + r0)) * CDIM, 16 * CDIM * 2);
#else
  for (int i = t; i < 16 * CDIM; i += 256)
    aT[i] = ptsH[((size_t)(b * NPTS + r0)) * CDIM + i];
#endif
  if (t < 16) sqRow[t] = sq[b * NPTS + r0 + t];
  __syncthreads();

  // A fragments (same 16 rows for every wave, held in registers all loop)
  const __bf16* arow = &aT[(lane & 15) * CDIM];
  v16bf a0 = load_tile(arow, lane, 0);
  v16bf a1 = load_tile(arow, lane, 1);

  float topV[KNN]; int topI[KNN];
#pragma unroll
  for (int i = 0; i < KNN; ++i) { topV[i] = NEG_INF; topI[i] = 0; }

  int myrow = t >> 4;                    // scan-phase: 16 threads per row
  int slot  = t & 15;
  int grow  = r0 + myrow;

  for (int it = 0; it < NPTS / 128; ++it) {       // 32 iterations
    int ct = it * 8 + wave;                       // this wave's column tile
    int c0 = ct * 16;
    const __bf16* brow = &ptsH[((size_t)(b * NPTS + c0 + (lane & 15))) * CDIM];
    v16bf b0 = load_tile(brow, lane, 0);
    v16bf b1 = load_tile(brow, lane, 1);
    v8f acc = {};
    acc = wmma_bf16(a0, b0, acc);
    acc = wmma_bf16(a1, b1, acc);
    float sqn = sq[b * NPTS + c0 + (lane & 15)];
#pragma unroll
    for (int r = 0; r < 8; ++r) {
      int m = r + ((lane >> 4) << 3);
      tileBuf[wave][m][lane & 15] = 2.f * acc[r] - sqRow[m] - sqn;
    }
    __syncthreads();
    // scan the 16x128 strip produced this iteration
    int colbase = it * 128;
#pragma unroll
    for (int j = 0; j < 8; ++j) {
      int cl   = slot * 8 + j;                    // 0..127
      int gcol = colbase + cl;
      float v = tileBuf[cl >> 4][myrow][cl & 15];
      if (gcol != grow && v > topV[KNN - 1]) {
        float cv = v; int ci = gcol;
#pragma unroll
        for (int s = 0; s < KNN; ++s) {
          if (cv > topV[s]) {
            float tv = topV[s]; topV[s] = cv; cv = tv;
            int   ti = topI[s]; topI[s] = ci; ci = ti;
          }
        }
      }
    }
    __syncthreads();
  }

#pragma unroll
  for (int s = 0; s < KNN; ++s) {
    candV[myrow][slot * KNN + s] = topV[s];
    candI[myrow][slot * KNN + s] = topI[s];
  }
  __syncthreads();
  if (t < 16) {
    int row = t;
    for (int s = 0; s < KNN; ++s) {
      float best = NEG_INF; int bj = 0;
      for (int j = 0; j < 16 * KNN; ++j) {
        float v = candV[row][j];
        if (v > best) { best = v; bj = j; }
      }
      candV[row][bj] = NEG_INF;
      idxOut[((size_t)(b * NPTS + r0 + row)) * KNN + s] = candI[row][bj];
    }
  }
}

// ---------------------------------------------------------------------------
// K2: fused per-point attention. 1 workgroup (8 waves) per point.
// Wave w: projection p=w>>2 (0=key from nbr, 1=value from nbr-q), head h=w&3.
// Padded 32x64 x 64x64 bf16 WMMA GEMMs; query/scores/softmax/output on VALU.
// ---------------------------------------------------------------------------
__global__ __launch_bounds__(256) void attn_kernel(
    const float* __restrict__ ptsF, const float* __restrict__ Wq,
    const __bf16* __restrict__ WkH, const __bf16* __restrict__ WvH,
    const int* __restrict__ idx, __bf16* __restrict__ mixH) {
  __shared__ float  qf[CDIM];
  __shared__ __bf16 nbrA[32 * CDIM];
  __shared__ __bf16 vA[32 * CDIM];
  __shared__ float  keyL[4][KNN][64];
  __shared__ float  valL[4][KNN][64];
  __shared__ float  queryL[4][64];
  __shared__ float  scoreL[4][KNN];

  int p = blockIdx.x;                    // flat (b,n)
  int bbase = p & ~(NPTS - 1);
  int t = threadIdx.x, lane = t & 31, wave = t >> 5;

  if (t < CDIM) qf[t] = ptsF[(size_t)p * CDIM + t];
  for (int i = t; i < 12 * CDIM; i += 256) {      // zero pad rows 20..31
    nbrA[KNN * CDIM + i] = (__bf16)0.f;
    vA[KNN * CDIM + i]   = (__bf16)0.f;
  }
  __syncthreads();
  for (int i = t; i < KNN * CDIM; i += 256) {
    int k = i >> 6, c = i & 63;
    int j = idx[(size_t)p * KNN + k];
    float nv = ptsF[((size_t)(bbase + j)) * CDIM + c];
    nbrA[i] = (__bf16)nv;
    vA[i]   = (__bf16)(nv - qf[c]);
  }
  __syncthreads();

  { // WMMA projections
    int proj = wave >> 2, h = wave & 3;
    const __bf16* W = (proj == 0 ? WkH : WvH) + (size_t)h * 64 * 64;
    const __bf16* A = (proj == 0 ? nbrA : vA);
    float* outL = (proj == 0 ? &keyL[h][0][0] : &valL[h][0][0]);
    for (int nt = 0; nt < 4; ++nt) {
      const __bf16* brow = W + (size_t)(nt * 16 + (lane & 15)) * 64;
      v16bf b0 = load_tile(brow, lane, 0);
      v16bf b1 = load_tile(brow, lane, 1);
      for (int mt = 0; mt < 2; ++mt) {
        const __bf16* ar = A + (size_t)(mt * 16 + (lane & 15)) * CDIM;
        v16bf a0 = load_tile(ar, lane, 0);
        v16bf a1 = load_tile(ar, lane, 1);
        v8f acc = {};
        acc = wmma_bf16(a0, b0, acc);
        acc = wmma_bf16(a1, b1, acc);
#pragma unroll
        for (int r = 0; r < 8; ++r) {
          int m = mt * 16 + r + ((lane >> 4) << 3);
          if (m < KNN) outL[m * 64 + nt * 16 + (lane & 15)] = acc[r];
        }
      }
    }
  }
  __syncthreads();

  { // query[h][e] = sum_c q[c] * Wq[h][e][c]
    int h = t >> 6, e = t & 63;
    const float* w = Wq + ((size_t)h * 64 + e) * 64;
    float s = 0.f;
#pragma unroll
    for (int c = 0; c < CDIM; ++c) s += qf[c] * w[c];
    queryL[h][e] = s;
  }
  __syncthreads();

  if (t < 4 * KNN) { // scores
    int h = t / KNN, k = t % KNN;
    float s = 0.f;
#pragma unroll
    for (int e = 0; e < 64; ++e) s += queryL[h][e] * keyL[h][k][e];
    scoreL[h][k] = s * 0.125f;           // / sqrt(64)
  }
  __syncthreads();
  if (t < 4) { // softmax over k
    int h = t;
    float mx = scoreL[h][0];
    for (int k = 1; k < KNN; ++k) mx = fmaxf(mx, scoreL[h][k]);
    float sum = 0.f;
    for (int k = 0; k < KNN; ++k) { float e = expf(scoreL[h][k] - mx); scoreL[h][k] = e; sum += e; }
    float inv = 1.f / sum;
    for (int k = 0; k < KNN; ++k) scoreL[h][k] *= inv;
  }
  __syncthreads();

  { // out[h][v] = sum_k s[k] * value[h][k][v]  -> mix (bf16 for next GEMM)
    int h = t >> 6, v = t & 63;
    float o = 0.f;
#pragma unroll
    for (int k = 0; k < KNN; ++k) o += scoreL[h][k] * valL[h][k][v];
    mixH[(size_t)p * 256 + h * 64 + v] = (__bf16)o;
  }
}

// ---------------------------------------------------------------------------
// K3: y[b][o][n] = sum_c Wout[o][c] * mix[b][n][c]. M=16 pts/block,
// 8 waves = 8 N-tiles of 16 (O=128), K=256 -> 8 chained WMMAs per wave.
// A-panel (16x256 bf16, 8KB contiguous) staged into LDS by the TDM.
// ---------------------------------------------------------------------------
__global__ __launch_bounds__(256) void outproj_kernel(
    const __bf16* __restrict__ mixH, const __bf16* __restrict__ WoutH,
    float* __restrict__ y) {
  __shared__ __bf16 aT[16 * 256];
  int p0 = blockIdx.x * 16;
  int t = threadIdx.x, lane = t & 31, wave = t >> 5;
#if HAVE_TDM
  if (t < 32) tdm_load_1d(aT, mixH + (size_t)p0 * 256, 16 * 256 * 2);
#else
  for (int i = t; i < 16 * 256; i += 256) aT[i] = mixH[(size_t)p0 * 256 + i];
#endif
  __syncthreads();
  const __bf16* arow = &aT[(lane & 15) * 256];
  const __bf16* brow = WoutH + (size_t)(wave * 16 + (lane & 15)) * 256;
  v8f acc = {};
#pragma unroll
  for (int kc = 0; kc < 8; ++kc) {
    v16bf a  = load_tile(arow, lane, kc);
    v16bf bb = load_tile(brow, lane, kc);
    acc = wmma_bf16(a, bb, acc);
  }
#pragma unroll
  for (int r = 0; r < 8; ++r) {
    int m = r + ((lane >> 4) << 3);
    int gp = p0 + m;
    int b = gp >> 12, n = gp & (NPTS - 1);
    int o = wave * 16 + (lane & 15);
    y[((size_t)b * 128 + o) * NPTS + n] = acc[r];
  }
}

// ---------------------------------------------------------------------------
// K4: BatchNorm stats per channel (population mean / rsqrt(var+eps))
// ---------------------------------------------------------------------------
__global__ __launch_bounds__(256) void bn_stats(
    const float* __restrict__ y, float* __restrict__ stats) {
  __shared__ float sb[256], s2b[256];
  int o = blockIdx.x;
  float s = 0.f, s2 = 0.f;
  for (int i = threadIdx.x; i < BATCH * NPTS; i += 256) {
    int b = i >> 12, n = i & (NPTS - 1);
    float v = y[((size_t)b * 128 + o) * NPTS + n];
    s += v; s2 += v * v;
  }
  sb[threadIdx.x] = s; s2b[threadIdx.x] = s2;
  __syncthreads();
  for (int st = 128; st > 0; st >>= 1) {
    if (threadIdx.x < st) { sb[threadIdx.x] += sb[threadIdx.x + st]; s2b[threadIdx.x] += s2b[threadIdx.x + st]; }
    __syncthreads();
  }
  if (threadIdx.x == 0) {
    float inv_n = 1.f / (float)(BATCH * NPTS);
    float mean = sb[0] * inv_n;
    float var  = s2b[0] * inv_n - mean * mean;
    stats[o] = mean;
    stats[128 + o] = rsqrtf(var + 1e-5f);
  }
}

// ---------------------------------------------------------------------------
// K5: normalize + affine + leaky-relu, concat with x -> out (B, 192, N)
// ---------------------------------------------------------------------------
__global__ __launch_bounds__(256) void finalize_kernel(
    const float* __restrict__ y, const float* __restrict__ x,
    const float* __restrict__ stats, const float* __restrict__ gamma,
    const float* __restrict__ beta, float* __restrict__ out) {
  size_t i = (size_t)blockIdx.x * 256 + threadIdx.x;
  const size_t total = (size_t)BATCH * 192 * NPTS;
  if (i >= total) return;
  int n   = (int)(i & (NPTS - 1));
  int ch  = (int)((i >> 12) % 192);
  int b   = (int)(i / ((size_t)192 * NPTS));
  float v;
  if (ch < 128) {
    float yy = y[((size_t)b * 128 + ch) * NPTS + n];
    v = (yy - stats[ch]) * stats[128 + ch] * gamma[ch] + beta[ch];
    v = (v >= 0.f) ? v : 0.2f * v;
  } else {
    v = x[((size_t)b * CDIM + (ch - 128)) * NPTS + n];
  }
  out[i] = v;
}

// ---------------------------------------------------------------------------
// Launch
// ---------------------------------------------------------------------------
static inline size_t align_up(size_t v) { return (v + 255) & ~(size_t)255; }

extern "C" void kernel_launch(void* const* d_in, const int* in_sizes, int n_in,
                              void* d_out, int out_size, void* d_ws, size_t ws_size,
                              hipStream_t stream) {
  const float* x     = (const float*)d_in[0];
  const float* Wq    = (const float*)d_in[1];
  const float* Wk    = (const float*)d_in[2];
  const float* Wv    = (const float*)d_in[3];
  const float* Wout  = (const float*)d_in[4];
  const float* gamma = (const float*)d_in[5];
  const float* beta  = (const float*)d_in[6];
  float* out = (float*)d_out;

  char* ws = (char*)d_ws;
  size_t off = 0;
  float*  ptsF  = (float*)(ws + off);  off = align_up(off + (size_t)BATCH * NPTS * CDIM * 4);
  __bf16* ptsH  = (__bf16*)(ws + off); off = align_up(off + (size_t)BATCH * NPTS * CDIM * 2);
  float*  sq    = (float*)(ws + off);  off = align_up(off + (size_t)BATCH * NPTS * 4);
  int*    idx   = (int*)(ws + off);    off = align_up(off + (size_t)BATCH * NPTS * KNN * 4);
  __bf16* WkH   = (__bf16*)(ws + off); off = align_up(off + (size_t)4 * 64 * 64 * 2);
  __bf16* WvH   = (__bf16*)(ws + off); off = align_up(off + (size_t)4 * 64 * 64 * 2);
  __bf16* WoutH = (__bf16*)(ws + off); off = align_up(off + (size_t)128 * 256 * 2);
  __bf16* mixH  = (__bf16*)(ws + off); off = align_up(off + (size_t)BATCH * NPTS * 256 * 2);
  float*  y     = (float*)(ws + off);  off = align_up(off + (size_t)BATCH * 128 * NPTS * 4);
  float*  stats = (float*)(ws + off);  off = align_up(off + (size_t)256 * 4);
  (void)ws_size; (void)in_sizes; (void)n_in; (void)out_size;

  prep_points<<<(BATCH * NPTS + 255) / 256, 256, 0, stream>>>(x, ptsF, ptsH, sq);
  conv_weights<<<(128 * 256 + 255) / 256, 256, 0, stream>>>(Wk, Wv, Wout, WkH, WvH, WoutH);
  knn_kernel<<<BATCH * (NPTS / 16), 256, 0, stream>>>(ptsH, sq, idx);
  attn_kernel<<<BATCH * NPTS, 256, 0, stream>>>(ptsF, Wq, WkH, WvH, idx, mixH);
  outproj_kernel<<<(BATCH * NPTS) / 16, 256, 0, stream>>>(mixH, WoutH, y);
  bn_stats<<<128, 256, 0, stream>>>(y, stats);
  finalize_kernel<<<(int)(((size_t)BATCH * 192 * NPTS + 255) / 256), 256, 0, stream>>>(
      y, x, stats, gamma, beta, out);
}
